// CLCBlock_39779987095768
// MI455X (gfx1250) — matrix operand order repaired and checked by reading, not compiled, not run on hardware
//
#include <hip/hip_runtime.h>
#include <cstdint>
#include <cstddef>

// ---------------------------------------------------------------------------
// Fused CLC block for MI455X (gfx1250, wave32):
//   stage kernel = [grouped 3x3 conv + BN + ReLU] -> LDS -> [grouped 1x1 conv
//   as WMMA GEMM + BN + ReLU] -> global.
// Launched twice (x->mid_f16 in ws, mid_f16->out fp32).
// BN is folded into conv weights/bias by tiny prep kernels; matrix math is
// f16 inputs with fp32 WMMA accumulation (v_wmma_f32_16x16x32_f16).
// Stage-2 input staging uses GLOBAL_LOAD_ASYNC_TO_LDS_B32 (ASYNCcnt path);
// stage-1 staging must pass through VGPRs (fp32 -> f16 conversion).
// Workspace needs ~51.5 MB (f16 intermediate + folded weights).
// ---------------------------------------------------------------------------

typedef _Float16 half_t;
typedef __attribute__((ext_vector_type(16))) _Float16 v16h;
typedef __attribute__((ext_vector_type(8)))  float    v8f;

#define B_    32
#define C_    256
#define H_    56
#define W_    56
#define TH    4            // tile rows per workgroup
#define XROW  60           // padded xs row: img col c -> xs col c+2 (zero halo)
#define XP    362          // xs per-channel pitch (6*60 + 2 pad; odd-dword stride)
#define MIDP  264          // mid per-pixel pitch (528B = 33*16: aligned b128, spread banks)
#define NPX   (TH * W_)    // 224 pixels per tile
#define EPSF  1e-5f

struct alignas(16) U4 { unsigned x, y, z, w; };
struct alignas(16) F4 { float x, y, z, w; };
struct Frag { U4 lo, hi; };   // 32B = one 16x32 f16 WMMA operand per lane

static __device__ inline unsigned pack2h(float a, float b) {
  union { _Float16 h[2]; unsigned u; } p;
  p.h[0] = (_Float16)a; p.h[1] = (_Float16)b;
  return p.u;
}

// load 8 consecutive fp32, produce 4 dwords of packed f16
static __device__ inline void load8f(const float* s, unsigned d[4]) {
  F4 a = *(const F4*)s;
  F4 b = *(const F4*)(s + 4);
  d[0] = pack2h(a.x, a.y); d[1] = pack2h(a.z, a.w);
  d[2] = pack2h(b.x, b.y); d[3] = pack2h(b.z, b.w);
}

static __device__ inline void store1(float*  p, float v) { *p = v; }
static __device__ inline void store1(half_t* p, float v) { *p = (half_t)v; }

// ---------------------------------------------------------------------------
// BN-folding prep kernels (one block of 256 threads each)
// ---------------------------------------------------------------------------
__global__ void fold_gconv(const float* __restrict__ w, const float* __restrict__ b,
                           const float* __restrict__ g, const float* __restrict__ bt,
                           const float* __restrict__ m, const float* __restrict__ v,
                           float* __restrict__ gwf) {
  const int o = threadIdx.x;                 // mid channel o = i*64 + k
  const float inv = g[o] * rsqrtf(v[o] + EPSF);
  const float* ws = w + o * 36;              // (i,k,c,kh,kw) flat = o*36 + c*9 + t
  for (int t = 0; t < 36; ++t) gwf[o * 40 + t] = ws[t] * inv;
  gwf[o * 40 + 36] = (b[o] - m[o]) * inv + bt[o];
}

__global__ void fold_pw(const float* __restrict__ w, const float* __restrict__ b,
                        const float* __restrict__ g, const float* __restrict__ bt,
                        const float* __restrict__ m, const float* __restrict__ v,
                        half_t* __restrict__ pwf, float* __restrict__ pbf) {
  const int q = threadIdx.x;
  const float inv = g[q] * rsqrtf(v[q] + EPSF);
  for (int c = 0; c < 64; ++c) pwf[q * 64 + c] = (half_t)(w[q * 64 + c] * inv);
  pbf[q] = (b[q] - m[q]) * inv + bt[q];
}

// ---------------------------------------------------------------------------
// Fused stage kernel
// ---------------------------------------------------------------------------
template <typename TIN, typename TOUT>
__global__ __launch_bounds__(512, 1)
void fused_stage(const TIN* __restrict__ in, const float* __restrict__ gwf,
                 const half_t* __restrict__ pwf, const float* __restrict__ pbf,
                 TOUT* __restrict__ out) {
  extern __shared__ char smem[];
  half_t* xs  = (half_t*)smem;                 // [C_][XP]  input tile (f16, zero halo)
  half_t* mid = xs + C_ * XP;                  // [NPX][MIDP] mid activation, px-major
  float*  pbl = (float*)(mid + NPX * MIDP);    // [C_] folded pw bias

  const int tid = threadIdx.x;
  const int n   = blockIdx.y;
  const int th0 = blockIdx.x * TH;

  // ---- stage-in: 256ch x 6 rows x 56 cols -> LDS f16, halo cols/rows zeroed
  for (int q = tid; q < C_; q += 512) pbl[q] = pbf[q];

  if constexpr (sizeof(TIN) == 2) {
    // f16 input: raw byte copy -> use async global->LDS DMA (ASYNCcnt path).
    // GVS addressing: SGPR64 base + per-lane 32-bit byte offset; per-lane LDS
    // destination address (low 32 bits of a generic LDS pointer == LDS offset).
    const unsigned long long base = (unsigned long long)(const void*)in;
    for (int i = tid; i < C_ * 6 * 28; i += 512) {
      const int ch  = i / 168;
      const int rem = i % 168;
      const int r   = rem / 28;
      const int m   = rem % 28;          // one dword = 2 f16 columns
      const int h   = th0 + r - 1;
      half_t* drow = xs + ch * XP + r * XROW;
      if (h >= 0 && h < H_) {
        const unsigned lds  = (unsigned)(uintptr_t)(drow + 2 + 2 * m);
        const unsigned goff = (unsigned)((((unsigned)(n * C_ + ch) * H_ + h) * W_ + 2 * m) * 2u);
        asm volatile("global_load_async_to_lds_b32 %0, %1, %2"
                     :: "v"(lds), "v"(goff), "s"(base) : "memory");
      } else {
        *(unsigned*)(drow + 2 + 2 * m) = 0u;   // zero-pad OOB rows
      }
      if (m == 0)  *(unsigned*)(drow)      = 0u;  // left halo cols
      if (m == 27) *(unsigned*)(drow + 58) = 0u;  // right halo cols
    }
    asm volatile("s_wait_asynccnt 0x0" ::: "memory");
  } else {
    // fp32 input: convert to f16 in VGPRs while staging.
    for (int i = tid; i < C_ * 6 * 7; i += 512) {
      const int ch  = i / 42;
      const int rem = i % 42;
      const int r   = rem / 7;
      const int j   = rem % 7;
      const int h   = th0 + r - 1;
      unsigned d[4] = {0u, 0u, 0u, 0u};
      if (h >= 0 && h < H_)
        load8f((const float*)in + ((size_t)(n * C_ + ch) * H_ + h) * W_ + 8 * j, d);
      unsigned* dst = (unsigned*)(xs + ch * XP + r * XROW + 2 + 8 * j);
      dst[0] = d[0]; dst[1] = d[1]; dst[2] = d[2]; dst[3] = d[3];
      if (j == 0) *(unsigned*)(xs + ch * XP + r * XROW)      = 0u;
      if (j == 6) *(unsigned*)(xs + ch * XP + r * XROW + 58) = 0u;
    }
  }
  __syncthreads();

  // ---- stage A: grouped 3x3 conv (BN folded) + ReLU -> mid (pixel-major f16)
  {
    const int o   = tid & 255;        // mid channel
    const int ph  = tid >> 8;         // which half of the 4 tile rows
    const int kin = (o & 63) * 4;     // first input channel of this depth group
    float wr[36], bias;
    {
      const F4* wp = (const F4*)(gwf + o * 40);
      #pragma unroll
      for (int q = 0; q < 9; ++q) {
        F4 t = wp[q];
        wr[4*q] = t.x; wr[4*q+1] = t.y; wr[4*q+2] = t.z; wr[4*q+3] = t.w;
      }
      bias = gwf[o * 40 + 36];
    }
    for (int rr = 0; rr < 2; ++rr) {
      const int rt = ph * 2 + rr;          // tile row 0..3
      for (int cj = 0; cj < 14; ++cj) {
        const int c0 = cj * 4;             // 4 output pixels per iteration
        float a0 = bias, a1 = bias, a2 = bias, a3 = bias;
        #pragma unroll
        for (int cc = 0; cc < 4; ++cc) {
          const half_t* xr0 = xs + (kin + cc) * XP + rt * XROW + (c0 + 1);
          #pragma unroll
          for (int dy = 0; dy < 3; ++dy) {
            const half_t* xr = xr0 + dy * XROW;
            const float x0 = (float)xr[0], x1 = (float)xr[1], x2 = (float)xr[2];
            const float x3 = (float)xr[3], x4 = (float)xr[4], x5 = (float)xr[5];
            const int wb = cc * 9 + dy * 3;
            a0 = fmaf(wr[wb], x0, a0); a0 = fmaf(wr[wb+1], x1, a0); a0 = fmaf(wr[wb+2], x2, a0);
            a1 = fmaf(wr[wb], x1, a1); a1 = fmaf(wr[wb+1], x2, a1); a1 = fmaf(wr[wb+2], x3, a1);
            a2 = fmaf(wr[wb], x2, a2); a2 = fmaf(wr[wb+1], x3, a2); a2 = fmaf(wr[wb+2], x4, a2);
            a3 = fmaf(wr[wb], x3, a3); a3 = fmaf(wr[wb+1], x4, a3); a3 = fmaf(wr[wb+2], x5, a3);
          }
        }
        const int p0 = rt * W_ + c0;
        mid[(p0+0)*MIDP + o] = (half_t)fmaxf(a0, 0.f);
        mid[(p0+1)*MIDP + o] = (half_t)fmaxf(a1, 0.f);
        mid[(p0+2)*MIDP + o] = (half_t)fmaxf(a2, 0.f);
        mid[(p0+3)*MIDP + o] = (half_t)fmaxf(a3, 0.f);
      }
    }
  }
  __syncthreads();

  // ---- stage B: grouped 1x1 conv via v_wmma_f32_16x16x32_f16, BN+ReLU, store
  {
    const int wv   = tid >> 5;        // wave id (uniform within wave)
    const int lane = tid & 31;
    const int lrow = lane & 15;
    const int lhi  = lane >> 4;
    for (int t = wv; t < 16 * 14; t += 16) {      // 16 M-tiles x 14 N-tiles, no divergence
      const int q0  = (t / 14) * 16;              // 16 output channels
      const int px0 = (t % 14) * 16;              // 16 pixels
      const int cb  = (q0 >> 6) * 64;             // pw group input-channel base
      const half_t* arow = pwf + (q0 + lrow) * 64;
      const half_t* brow = mid + (px0 + lrow) * MIDP + cb;
      v8f acc = {};
      #pragma unroll
      for (int ks = 0; ks < 2; ++ks) {
        Frag fa, fb;
        // A (16x32): lane<16 K={0..7,16..23}, lane>=16 K={8..15,24..31}
        fa.lo = *(const U4*)(arow + ks * 32 + lhi * 8);
        fa.hi = *(const U4*)(arow + ks * 32 + lhi * 8 + 16);
        // B (32x16): lane<16 K={0..7,16..23}, lane>=16 K={8..15,24..31}; col = lane&15
        fb.lo = *(const U4*)(brow + ks * 32 + lhi * 8);
        fb.hi = *(const U4*)(brow + ks * 32 + lhi * 8 + 16);
        acc = __builtin_amdgcn_wmma_f32_16x16x32_f16(
                false, __builtin_bit_cast(v16h, fa),
                false, __builtin_bit_cast(v16h, fb),
                (short)0, acc, false, false);
      }
      const int p  = px0 + lrow;       // N = lane&15
      const int hh = th0 + p / W_;
      const int wc = p % W_;
      #pragma unroll
      for (int r = 0; r < 8; ++r) {    // C/D: vgpr r -> M = r + 8*(lane>=16)
        const int q = q0 + lhi * 8 + r;
        const float vv = fmaxf(acc[r] + pbl[q], 0.f);
        store1(out + ((size_t)(n * C_ + q) * H_ + hh) * W_ + wc, vv);
      }
    }
  }
}

// ---------------------------------------------------------------------------
extern "C" void kernel_launch(void* const* d_in, const int* in_sizes, int n_in,
                              void* d_out, int out_size, void* d_ws, size_t ws_size,
                              hipStream_t stream) {
  (void)in_sizes; (void)n_in; (void)out_size; (void)ws_size;

  const float* x     = (const float*)d_in[0];
  const float* w1    = (const float*)d_in[1];
  const float* b1    = (const float*)d_in[2];
  const float* bn1ag = (const float*)d_in[3];
  const float* bn1ab = (const float*)d_in[4];
  const float* bn1am = (const float*)d_in[5];
  const float* bn1av = (const float*)d_in[6];
  const float* pw1   = (const float*)d_in[7];
  const float* pb1   = (const float*)d_in[8];
  const float* bn1bg = (const float*)d_in[9];
  const float* bn1bb = (const float*)d_in[10];
  const float* bn1bm = (const float*)d_in[11];
  const float* bn1bv = (const float*)d_in[12];
  const float* w2    = (const float*)d_in[13];
  const float* b2    = (const float*)d_in[14];
  const float* bn2ag = (const float*)d_in[15];
  const float* bn2ab = (const float*)d_in[16];
  const float* bn2am = (const float*)d_in[17];
  const float* bn2av = (const float*)d_in[18];
  const float* pw2   = (const float*)d_in[19];
  const float* pb2   = (const float*)d_in[20];
  const float* bn2bg = (const float*)d_in[21];
  const float* bn2bb = (const float*)d_in[22];
  const float* bn2bm = (const float*)d_in[23];
  const float* bn2bv = (const float*)d_in[24];

  // workspace carve (all 16B-aligned)
  char* ws = (char*)d_ws;
  size_t off = 0;
  half_t* midbuf = (half_t*)(ws + off); off += (size_t)B_ * C_ * H_ * W_ * sizeof(half_t);
  float*  gwf1   = (float*) (ws + off); off += 256 * 40 * sizeof(float);
  half_t* pwf1   = (half_t*)(ws + off); off += 256 * 64 * sizeof(half_t);
  float*  pbf1   = (float*) (ws + off); off += 256 * sizeof(float);
  float*  gwf2   = (float*) (ws + off); off += 256 * 40 * sizeof(float);
  half_t* pwf2   = (half_t*)(ws + off); off += 256 * 64 * sizeof(half_t);
  float*  pbf2   = (float*) (ws + off); off += 256 * sizeof(float);

  fold_gconv<<<1, 256, 0, stream>>>(w1, b1, bn1ag, bn1ab, bn1am, bn1av, gwf1);
  fold_pw   <<<1, 256, 0, stream>>>(pw1, pb1, bn1bg, bn1bb, bn1bm, bn1bv, pwf1, pbf1);
  fold_gconv<<<1, 256, 0, stream>>>(w2, b2, bn2ag, bn2ab, bn2am, bn2av, gwf2);
  fold_pw   <<<1, 256, 0, stream>>>(pw2, pb2, bn2bg, bn2bb, bn2bm, bn2bv, pwf2, pbf2);

  const size_t smem = (size_t)C_ * XP * sizeof(half_t)
                    + (size_t)NPX * MIDP * sizeof(half_t)
                    + (size_t)C_ * sizeof(float);   // ~298 KB < 320 KB/WGP

  hipFuncSetAttribute(reinterpret_cast<const void*>(&fused_stage<float, half_t>),
                      hipFuncAttributeMaxDynamicSharedMemorySize, (int)smem);
  hipFuncSetAttribute(reinterpret_cast<const void*>(&fused_stage<half_t, float>),
                      hipFuncAttributeMaxDynamicSharedMemorySize, (int)smem);

  dim3 grid(H_ / TH, B_);   // 14 x 32 workgroups, 512 threads (16 wave32) each
  fused_stage<float, half_t><<<grid, 512, smem, stream>>>(x,      gwf1, pwf1, pbf1, midbuf);
  fused_stage<half_t, float><<<grid, 512, smem, stream>>>(midbuf, gwf2, pwf2, pbf2, (float*)d_out);
}
